// RandLANet_82085414961513
// MI455X (gfx1250) — matrix operand order repaired and checked by reading, not compiled, not run on hardware
//
#include <hip/hip_runtime.h>
#include <hip/hip_bf16.h>

// ---------------------------------------------------------------------------
// RandLA-Net forward for MI455X (gfx1250, wave32, WMMA).
// All linear layers run through a v_wmma_f32_16x16x32_f16 GEMM whose operands
// are pre-converted into zero-padded f16 staging buffers so the inner loop is
// 4x global_load_b128 + 1x v_wmma with no branches or converts.
// ---------------------------------------------------------------------------

#define BATCH 2
#define NPTS  8192
#define KNN_K 16

typedef __attribute__((ext_vector_type(16))) _Float16 v16h;
typedef __attribute__((ext_vector_type(8)))  _Float16 v8h;
typedef __attribute__((ext_vector_type(8)))  float    v8f;

#define ACT_NONE     0
#define ACT_LRELU02  1
#define ACT_LRELU001 2
#define ACT_RELU     3

__device__ __forceinline__ float act_f(float v, int act) {
  if (act == ACT_LRELU02)  return v > 0.f ? v : 0.2f  * v;
  if (act == ACT_LRELU001) return v > 0.f ? v : 0.01f * v;
  if (act == ACT_RELU)     return v > 0.f ? v : 0.f;
  return v;
}

// ---------------------------------------------------------------------------
// Convert + zero-pad f32 (rows x Ci) -> f16 (rows_p x Kp).
// Also used for weights (rows = Co, rows_p = Co_p).
// ---------------------------------------------------------------------------
__global__ void cvt_pad_kernel(const float* __restrict__ src,
                               _Float16* __restrict__ dst,
                               int rows, int Ci, int Kp, long long total)
{
  long long t = (long long)blockIdx.x * blockDim.x + threadIdx.x;
  if (t >= total) return;
  int k = (int)(t % Kp);
  long long r = t / Kp;
  dst[t] = (r < rows && k < Ci) ? (_Float16)src[r * Ci + k] : (_Float16)0.f;
}

// ---------------------------------------------------------------------------
// GEMM: Y[rows x Co] = act( Xh[rows_p x Kp] * Wh[Co_p x Kp]^T  (+ epilogue) )
// 4 waves per block, one 16x16 tile per wave.
// A fragment per lane: two contiguous 8-half runs  (16B + 16B loads).
// B fragment per lane: two contiguous 8-half runs  (16B + 16B loads).
// ---------------------------------------------------------------------------
__global__ __launch_bounds__(128)
void gemm_wmma_kernel(const _Float16* __restrict__ Xh,
                      const _Float16* __restrict__ Wh,
                      const float* __restrict__ bias,
                      const float* __restrict__ gamma, const float* __restrict__ beta,
                      const float* __restrict__ residual,
                      float* __restrict__ Y, int ldy,
                      int rows, int Co, int Kp, int rows_p, int act)
{
  const int wid  = threadIdx.x >> 5;
  const int lane = threadIdx.x & 31;
  const int half = lane >> 4;       // 0: lanes 0-15, 1: lanes 16-31
  const int l15  = lane & 15;
  const int tm   = (blockIdx.x * 4 + wid) * 16;
  const int tn   = blockIdx.y * 16;
  if (tm >= rows_p) return;         // wave-uniform: EXEC stays all-ones

  const int ncol = tn + l15;

  // hoisted, branch-free operand pointers (all 16-byte aligned)
  const _Float16* ap = Xh + (size_t)(tm + l15) * Kp + (half ? 8 : 0);
  const _Float16* bp = Wh + (size_t)ncol * Kp + (half ? 8 : 0);

  v8f acc;
  float binit = (bias != nullptr && ncol < Co) ? bias[ncol] : 0.f;
#pragma unroll
  for (int i = 0; i < 8; ++i) acc[i] = binit;

  for (int k0 = 0; k0 < Kp; k0 += 32) {
    v8h a0 = *(const v8h*)(ap);
    v8h a1 = *(const v8h*)(ap + 16);
    v8h b0 = *(const v8h*)(bp);
    v8h b1 = *(const v8h*)(bp + 16);
    v16h a = __builtin_shufflevector(a0, a1, 0, 1, 2, 3, 4, 5, 6, 7,
                                     8, 9, 10, 11, 12, 13, 14, 15);
    v16h b = __builtin_shufflevector(b0, b1, 0, 1, 2, 3, 4, 5, 6, 7,
                                     8, 9, 10, 11, 12, 13, 14, 15);
    acc = __builtin_amdgcn_wmma_f32_16x16x32_f16(
        /*neg_a=*/false, a, /*neg_b=*/false, b,
        /*c_mod=*/(short)0, acc, /*reuse_a=*/false, /*reuse_b=*/false);
    ap += 32;
    bp += 32;
  }

  if (ncol < Co) {
    // batchnorm affine: x / sqrt(1 + 1e-6) * g + b
    const float invs = 0.99999950f;
    float gn = 1.f, bt = 0.f;
    const bool has_bn = (gamma != nullptr);
    if (has_bn) { gn = gamma[ncol] * invs; bt = beta[ncol]; }
#pragma unroll
    for (int i = 0; i < 8; ++i) {
      const int mo = tm + (half ? 8 : 0) + i;
      if (mo < rows) {
        float v = acc[i];
        if (has_bn) v = v * gn + bt;
        if (residual) v += residual[(size_t)mo * Co + ncol];
        Y[(size_t)mo * ldy + ncol] = act_f(v, act);
      }
    }
  }
}

// ---------------------------------------------------------------------------
// coords[b, i, :] = points[b, perm[i], :]
// ---------------------------------------------------------------------------
__global__ void permute_kernel(const float* __restrict__ pts,
                               const int* __restrict__ perm,
                               float* __restrict__ coords, int n)
{
  int t = blockIdx.x * blockDim.x + threadIdx.x;
  if (t >= BATCH * n * 3) return;
  int c = t % 3;
  int i = (t / 3) % n;
  int b = t / (3 * n);
  coords[((size_t)b * n + i) * 3 + c] = pts[((size_t)b * n + perm[i]) * 3 + c];
}

// ---------------------------------------------------------------------------
// Per-query top-16 nearest neighbors (squared euclidean, stable ties).
// ---------------------------------------------------------------------------
__global__ void knn_topk_kernel(const float* __restrict__ coords, int n,
                                int bstride, int* __restrict__ idx_out)
{
  int t = blockIdx.x * blockDim.x + threadIdx.x;
  if (t >= BATCH * n) return;
  int b = t / n, i = t % n;
  const float* cb = coords + (size_t)b * bstride;
  float qx = cb[3 * i], qy = cb[3 * i + 1], qz = cb[3 * i + 2];
  float dd[KNN_K];
  int   ii[KNN_K];
#pragma unroll
  for (int k = 0; k < KNN_K; ++k) { dd[k] = 3.4e38f; ii[k] = 0; }
  for (int j = 0; j < n; ++j) {
    if (j + 32 < n) __builtin_prefetch(cb + 3 * (j + 32), 0, 0);
    float dx = qx - cb[3 * j];
    float dy = qy - cb[3 * j + 1];
    float dz = qz - cb[3 * j + 2];
    float d2 = dx * dx + dy * dy + dz * dz;
    if (d2 < dd[KNN_K - 1]) {
      int pos = KNN_K - 1;
      while (pos > 0 && dd[pos - 1] > d2) {
        dd[pos] = dd[pos - 1]; ii[pos] = ii[pos - 1]; --pos;
      }
      dd[pos] = d2; ii[pos] = j;
    }
  }
  int* o = idx_out + (size_t)t * KNN_K;
#pragma unroll
  for (int k = 0; k < KNN_K; ++k) o[k] = ii[k];
}

// Decoder 1-NN: nearest source point for every up-sampled point.
__global__ void knn1_kernel(const float* __restrict__ coords, int nup, int nsrc,
                            int bstride, int* __restrict__ idx_out)
{
  int t = blockIdx.x * blockDim.x + threadIdx.x;
  if (t >= BATCH * nup) return;
  int b = t / nup, i = t % nup;
  const float* cb = coords + (size_t)b * bstride;
  float qx = cb[3 * i], qy = cb[3 * i + 1], qz = cb[3 * i + 2];
  float best = 3.4e38f; int bi = 0;
  for (int j = 0; j < nsrc; ++j) {
    float dx = qx - cb[3 * j];
    float dy = qy - cb[3 * j + 1];
    float dz = qz - cb[3 * j + 2];
    float d2 = dx * dx + dy * dy + dz * dz;
    if (d2 < best) { best = d2; bi = j; }
  }
  idx_out[t] = bi;
}

// ---------------------------------------------------------------------------
// Relative point-position encoding: [orig(3), neigh(3), rel(3), |rel|] -> 10
// ---------------------------------------------------------------------------
__global__ void build_enc_kernel(const float* __restrict__ coords, int n,
                                 int bstride, const int* __restrict__ idx,
                                 float* __restrict__ enc)
{
  int t = blockIdx.x * blockDim.x + threadIdx.x;
  if (t >= BATCH * n * KNN_K) return;
  int i = (t / KNN_K) % n;
  int b = t / (KNN_K * n);
  const float* cb = coords + (size_t)b * bstride;
  float ox = cb[3 * i], oy = cb[3 * i + 1], oz = cb[3 * i + 2];
  int j = idx[t];
  float nx = cb[3 * j], ny = cb[3 * j + 1], nz = cb[3 * j + 2];
  float rx = ox - nx, ry = oy - ny, rz = oz - nz;
  float d = sqrtf(rx * rx + ry * ry + rz * rz);
  float* e = enc + (size_t)t * 10;
  e[0] = ox; e[1] = oy; e[2] = oz;
  e[3] = nx; e[4] = ny; e[5] = nz;
  e[6] = rx; e[7] = ry; e[8] = rz;
  e[9] = d;
}

// Broadcast per-point features into second half of the LSE tensor.
__global__ void fill_half_kernel(const float* __restrict__ feats, int Cf,
                                 float* __restrict__ lse, int C2, int hoff,
                                 long long total)
{
  long long t = (long long)blockIdx.x * blockDim.x + threadIdx.x;
  if (t >= total) return;
  int c = (int)(t % Cf);
  long long pk = t / Cf;          // p*K + k
  long long p  = pk / KNN_K;
  lse[pk * C2 + hoff + c] = feats[p * Cf + c];
}

// Attention pooling: softmax over K per channel, weighted sum.
__global__ void attpool_kernel(const float* __restrict__ x, int C,
                               float* __restrict__ pooled, int rows)
{
  int t = blockIdx.x * blockDim.x + threadIdx.x;
  if (t >= rows * C) return;
  int p = t / C, c = t % C;
  const float* xb = x + ((size_t)p * KNN_K) * C + c;
  float m = -3.4e38f;
#pragma unroll
  for (int k = 0; k < KNN_K; ++k) m = fmaxf(m, xb[(size_t)k * C]);
  float se = 0.f, ws = 0.f;
#pragma unroll
  for (int k = 0; k < KNN_K; ++k) {
    float xv = xb[(size_t)k * C];
    float e = __expf(xv - m);
    se += e; ws += e * xv;
  }
  pooled[t] = ws / se;
}

// Random-downsample: keep first n_new rows of every batch.
__global__ void compact_kernel(const float* __restrict__ src, int n_old,
                               int n_new, int C, float* __restrict__ dst)
{
  int t = blockIdx.x * blockDim.x + threadIdx.x;
  if (t >= BATCH * n_new * C) return;
  int c = t % C;
  int r = (t / C) % n_new;
  int b = t / (C * n_new);
  dst[t] = src[((size_t)b * n_old + r) * C + c];
}

// Decoder: dst[b,r,:] = concat( x[b, nn[b,r], :Cx], skip[b, r, :Cs] )
__global__ void gather_concat_kernel(const float* __restrict__ xsrc, int nsrc,
                                     int Cx, const int* __restrict__ nn,
                                     const float* __restrict__ skip, int Cs,
                                     float* __restrict__ dst, int nup)
{
  int Cd = Cx + Cs;
  long long t = (long long)blockIdx.x * blockDim.x + threadIdx.x;
  if (t >= (long long)BATCH * nup * Cd) return;
  int c = (int)(t % Cd);
  int r = (int)((t / Cd) % nup);
  int b = (int)(t / ((long long)Cd * nup));
  size_t p = (size_t)b * nup + r;
  if (c < Cx) dst[t] = xsrc[((size_t)b * nsrc + nn[p]) * Cx + c];
  else        dst[t] = skip[p * Cs + (c - Cx)];
}

// ---------------------------------------------------------------------------
// Host-side orchestration
// ---------------------------------------------------------------------------
static inline int cdiv(long long a, long long b) { return (int)((a + b - 1) / b); }

static _Float16* g_xh = nullptr;
static _Float16* g_wh = nullptr;

static void launch_gemm(const float* X, const float* W, const float* bias,
                        const float* g, const float* bt, const float* res,
                        float* Y, int ldy, int rows, int Ci, int Co, int act,
                        hipStream_t s)
{
  const int Kp     = (Ci + 31) & ~31;
  const int rows_p = (rows + 15) & ~15;
  const int Co_p   = (Co + 15) & ~15;
  const long long xtot = (long long)rows_p * Kp;
  const long long wtot = (long long)Co_p * Kp;
  cvt_pad_kernel<<<cdiv(xtot, 256), 256, 0, s>>>(X, g_xh, rows, Ci, Kp, xtot);
  cvt_pad_kernel<<<cdiv(wtot, 256), 256, 0, s>>>(W, g_wh, Co, Ci, Kp, wtot);
  dim3 grid(cdiv(rows_p / 16, 4), Co_p / 16);
  gemm_wmma_kernel<<<grid, dim3(128), 0, s>>>(g_xh, g_wh, bias, g, bt, res,
                                              Y, ldy, rows, Co, Kp, rows_p, act);
}

// workspace layout (float offsets)
static const size_t OFF_COORDS = 0;                  //  2*8192*3   =   49152
static const size_t OFF_FA  = OFF_COORDS + 49152;    //  1048576
static const size_t OFF_FB  = OFF_FA  + 1048576;     //  1048576
static const size_t OFF_FC  = OFF_FB  + 1048576;     //  1048576
static const size_t OFF_SK1 = OFF_FC  + 1048576;     //  2*8192*32  =  524288
static const size_t OFF_SK2 = OFF_SK1 + 524288;      //  2*2048*128 =  524288
static const size_t OFF_SK3 = OFF_SK2 + 524288;      //  2*512*256  =  262144
static const size_t OFF_SK4 = OFF_SK3 + 262144;      //  2*128*512  =  131072
static const size_t OFF_ENC = OFF_SK4 + 131072;      //  2*8192*16*10 = 2621440
static const size_t OFF_LSE = OFF_ENC + 2621440;     //  2*8192*16*16 = 4194304
static const size_t OFF_POOL= OFF_LSE + 4194304;     //  262144
static const size_t OFF_SHT = OFF_POOL+ 262144;      //  524288
static const size_t OFF_IDX = OFF_SHT + 524288;      //  262144 (ints)
static const size_t OFF_XH  = OFF_IDX + 262144;      //  4194304 floats = 8.4M halfs
static const size_t OFF_WH  = OFF_XH  + 4194304;     //  131072 floats  = 262K halfs

extern "C" void kernel_launch(void* const* d_in, const int* in_sizes, int n_in,
                              void* d_out, int out_size, void* d_ws, size_t ws_size,
                              hipStream_t stream)
{
  (void)in_sizes; (void)n_in; (void)out_size; (void)ws_size;

  // ---- inputs (setup_inputs() flattening order) ----
  const float* points = (const float*)d_in[0];
  const int*   perm   = (const int*)d_in[1];
  const float* fcw  = (const float*)d_in[2];
  const float* fcb  = (const float*)d_in[3];
  const float* bn0g = (const float*)d_in[4];
  const float* bn0b = (const float*)d_in[5];
  // encoder levels: 16 tensors each
  // 0 m1w 1 m1b 2 l1w 3 l1b 4 p1w 5 p1b 6 l2w 7 l2b
  // 8 p2w 9 p2b 10 m2w 11 m2b 12 m3w 13 m3b 14 bng 15 bnb
  const float* ep[4][16];
  int ix = 6;
  for (int l = 0; l < 4; ++l)
    for (int t = 0; t < 16; ++t) ep[l][t] = (const float*)d_in[ix++];
  const float* dw[4]; const float* db[4];
  for (int d = 0; d < 4; ++d) {
    dw[d] = (const float*)d_in[ix++];
    db[d] = (const float*)d_in[ix++];
  }
  const float* mw  = (const float*)d_in[ix++];
  const float* mb  = (const float*)d_in[ix++];
  const float* f1w = (const float*)d_in[ix++];
  const float* f1b = (const float*)d_in[ix++];
  const float* f2w = (const float*)d_in[ix++];
  const float* f2b = (const float*)d_in[ix++];
  const float* oww = (const float*)d_in[ix++];
  const float* obb = (const float*)d_in[ix++];

  // ---- workspace carve-up ----
  float* ws     = (float*)d_ws;
  float* coords = ws + OFF_COORDS;
  float* fA     = ws + OFF_FA;
  float* fB     = ws + OFF_FB;
  float* fC     = ws + OFF_FC;
  float* sk[4]  = { ws + OFF_SK1, ws + OFF_SK2, ws + OFF_SK3, ws + OFF_SK4 };
  float* encb   = ws + OFF_ENC;
  float* lseb   = ws + OFF_LSE;
  float* pool   = ws + OFF_POOL;
  float* shrt   = ws + OFF_SHT;
  int*   idxb   = (int*)(ws + OFF_IDX);
  g_xh          = (_Float16*)(ws + OFF_XH);
  g_wh          = (_Float16*)(ws + OFF_WH);

  const int BS = NPTS * 3;   // coords batch stride (floats)

  // ---- input embedding: permute, fc(3->8) + bn0 + leaky(0.2) ----
  permute_kernel<<<cdiv(BATCH * NPTS * 3, 256), 256, 0, stream>>>(points, perm,
                                                                  coords, NPTS);
  launch_gemm(coords, fcw, fcb, bn0g, bn0b, nullptr,
              fA, 8, BATCH * NPTS, 3, 8, ACT_LRELU02, stream);

  // ---- encoder ----
  const int nlv[4]   = {8192, 2048, 512, 128};
  const int dinv[4]  = {8, 32, 128, 256};
  const int doutv[4] = {16, 64, 128, 256};
  for (int l = 0; l < 4; ++l) {
    const int n = nlv[l], din = dinv[l], dO = doutv[l];
    const int h = dO / 2, d2o = 2 * dO;
    const int rows = BATCH * n;
    const long long rkh = (long long)rows * KNN_K * h;

    knn_topk_kernel<<<cdiv(rows, 128), 128, 0, stream>>>(coords, n, BS, idxb);
    // m1: din -> h, leaky 0.2
    launch_gemm(fA, ep[l][0], ep[l][1], nullptr, nullptr, nullptr,
                fB, h, rows, din, h, ACT_LRELU02, stream);
    // relative position encoding (shared between both LSEs)
    build_enc_kernel<<<cdiv(rows * KNN_K, 256), 256, 0, stream>>>(coords, n, BS,
                                                                  idxb, encb);
    // LSE 1: lin(enc) -> [:,0:h], broadcast feats -> [:,h:2h]
    launch_gemm(encb, ep[l][2], ep[l][3], nullptr, nullptr, nullptr,
                lseb, dO, rows * KNN_K, 10, h, ACT_NONE, stream);
    fill_half_kernel<<<cdiv(rkh, 256), 256, 0, stream>>>(fB, h, lseb, dO, h, rkh);
    // attpool 1 (softmax over K) + p1: dO -> h
    attpool_kernel<<<cdiv(rows * dO, 256), 256, 0, stream>>>(lseb, dO, pool, rows);
    launch_gemm(pool, ep[l][4], ep[l][5], nullptr, nullptr, nullptr,
                fC, h, rows, dO, h, ACT_NONE, stream);
    // LSE 2 + attpool 2 + p2: dO -> dO
    launch_gemm(encb, ep[l][6], ep[l][7], nullptr, nullptr, nullptr,
                lseb, dO, rows * KNN_K, 10, h, ACT_NONE, stream);
    fill_half_kernel<<<cdiv(rkh, 256), 256, 0, stream>>>(fC, h, lseb, dO, h, rkh);
    attpool_kernel<<<cdiv(rows * dO, 256), 256, 0, stream>>>(lseb, dO, pool, rows);
    launch_gemm(pool, ep[l][8], ep[l][9], nullptr, nullptr, nullptr,
                fC, dO, rows, dO, dO, ACT_NONE, stream);
    // shortcut = bn(m3(feats)); out = leaky0.01(m2(x) + shortcut) -> skip[l]
    launch_gemm(fA, ep[l][12], ep[l][13], ep[l][14], ep[l][15], nullptr,
                shrt, d2o, rows, din, d2o, ACT_NONE, stream);
    launch_gemm(fC, ep[l][10], ep[l][11], nullptr, nullptr, shrt,
                sk[l], d2o, rows, dO, d2o, ACT_LRELU001, stream);
    // random downsample (take first n/4 rows per batch)
    const int nn_ = n / 4;
    compact_kernel<<<cdiv(BATCH * nn_ * d2o, 256), 256, 0, stream>>>(
        sk[l], n, nn_, d2o, fA);
  }

  // ---- bottleneck: relu(mw @ x) at n=32, C=512 ----
  launch_gemm(fA, mw, mb, nullptr, nullptr, nullptr,
              fB, 512, BATCH * 32, 512, 512, ACT_RELU, stream);

  // ---- decoder ----
  const int nupv[4] = {128, 512, 2048, 8192};
  const int csv[4]  = {512, 256, 128, 32};    // skip channels (popped order)
  const int cov[4]  = {256, 128, 32, 8};
  float* skp[4] = { sk[3], sk[2], sk[1], sk[0] };
  float* xcur = fB;
  float* othr = fA;
  int ncur = 32, Cx = 512;
  for (int d = 0; d < 4; ++d) {
    const int nup = nupv[d], Cs = csv[d], Co = cov[d];
    const int Cin = Cx + Cs;
    knn1_kernel<<<cdiv(BATCH * nup, 128), 128, 0, stream>>>(coords, nup, ncur,
                                                            BS, idxb);
    gather_concat_kernel<<<cdiv((long long)BATCH * nup * Cin, 256), 256, 0,
                          stream>>>(xcur, ncur, Cx, idxb, skp[d], Cs, fC, nup);
    launch_gemm(fC, dw[d], db[d], nullptr, nullptr, nullptr,
                othr, Co, BATCH * nup, Cin, Co, ACT_NONE, stream);
    float* tmp = xcur; xcur = othr; othr = tmp;
    ncur = nup; Cx = Co;
  }

  // ---- head: f1(8->64) relu, f2(64->32) relu, ow(32->13) ----
  launch_gemm(xcur, f1w, f1b, nullptr, nullptr, nullptr,
              fC, 64, BATCH * NPTS, 8, 64, ACT_RELU, stream);
  launch_gemm(fC, f2w, f2b, nullptr, nullptr, nullptr,
              fA, 32, BATCH * NPTS, 64, 32, ACT_RELU, stream);
  launch_gemm(fA, oww, obb, nullptr, nullptr, nullptr,
              (float*)d_out, 13, BATCH * NPTS, 32, 13, ACT_NONE, stream);
}
